// Qwen3Attention_64742337020184
// MI455X (gfx1250) — matrix operand order repaired and checked by reading, not compiled, not run on hardware
//
#include <hip/hip_runtime.h>
#include <hip/hip_bf16.h>
#include <cstdint>
#include <cstddef>

// ---------------- problem constants ----------------
#define HQ    16
#define HKVN  8
#define DH    128
#define HID   2048
#define BATCH 2
#define SEQ   2048
#define EPSV  1e-5f
#define SCALEV 0.08838834764831843f   // 128^-0.5
#define LOGBASE 9.210340371976184f    // ln(10000)

typedef __attribute__((ext_vector_type(16))) __bf16 v16bf;
typedef __attribute__((ext_vector_type(8)))  float  v8f;
typedef int v4i __attribute__((ext_vector_type(4)));
typedef unsigned int v4u __attribute__((ext_vector_type(4)));
typedef v4u __attribute__((may_alias)) v4u_a;
typedef unsigned int v2u __attribute__((ext_vector_type(2)));
typedef v2u __attribute__((may_alias)) v2u_a;

union FragAB { v16bf bf; unsigned int u[8]; };

__device__ __forceinline__ unsigned short f2bf(float f) {
  unsigned int u = __float_as_uint(f);
  unsigned int r = u + 0x7FFFu + ((u >> 16) & 1u);   // round-to-nearest-even
  return (unsigned short)(r >> 16);
}

// ---------------- CDNA5 async global->LDS staging (ASYNCcnt path) ----------------
#if defined(__AMDGCN__) && __has_builtin(__builtin_amdgcn_global_load_async_to_lds_b128)
#define HAS_ASYNC 1
#else
#define HAS_ASYNC 0
#endif

#if HAS_ASYNC
typedef __attribute__((address_space(1))) v4i* gv4i_p;   // global int4*
typedef __attribute__((address_space(3))) v4i* lv4i_p;   // LDS int4*
#endif

__device__ __forceinline__ void stage16(const unsigned short* g, unsigned short* l) {
#if HAS_ASYNC
  gv4i_p gp = (gv4i_p)(v4i*)const_cast<unsigned short*>(g);
  lv4i_p lp = (lv4i_p)(v4i*)l;
  __builtin_amdgcn_global_load_async_to_lds_b128(gp, lp, 0, 0);
#else
  *(v4u_a*)l = *(const v4u_a*)g;   // global_load_b128 + ds_store_b128
#endif
}

// Wait until at most N async stage ops remain in flight (in-order completion),
// enabling double-buffered pipelining of global->LDS tile copies.
template <int N>
__device__ __forceinline__ void wait_stage() {
#if HAS_ASYNC
#if __has_builtin(__builtin_amdgcn_s_wait_asynccnt)
  __builtin_amdgcn_s_wait_asynccnt(N);
#else
  asm volatile("s_wait_asynccnt %0" :: "i"(N) : "memory");
#endif
#endif
}

// ---------------- fp32 -> bf16 conversion (vectorized x4) ----------------
__global__ __launch_bounds__(256) void cvt_f32_bf16(const float* __restrict__ in,
                                                    unsigned short* __restrict__ out,
                                                    int n) {
  int i = (int)(blockIdx.x * blockDim.x + threadIdx.x) << 2;
  if (i >= n) return;
  float4 f = *(const float4*)(in + i);
  v2u p;
  p.x = (unsigned)f2bf(f.x) | ((unsigned)f2bf(f.y) << 16);
  p.y = (unsigned)f2bf(f.z) | ((unsigned)f2bf(f.w) << 16);
  *(v2u_a*)(out + i) = p;
}

// ---------------- bf16 WMMA GEMM: C[M,N] = A[M,K] * B[N,K]^T ----------------
// Block tile 128x64, K-step 32, 256 threads = 8 wave32s, each wave: 16 rows x 64 cols.
// Double-buffered async global->LDS staging (3 async b128 ops per thread per tile).
#define TM 128
#define TN 64
#define TK 32
#define LDT (TK + 8)   // padded LDS stride (ushorts): 80B rows, 16B-aligned chunks

__global__ __launch_bounds__(256)
void gemm_bf16_f32(const unsigned short* __restrict__ A,
                   const unsigned short* __restrict__ Bm,
                   float* __restrict__ C, int M, int N, int K) {
  __shared__ __align__(16) unsigned short As[2][TM * LDT];
  __shared__ __align__(16) unsigned short Bs[2][TN * LDT];
  const int tid  = threadIdx.x;
  const int lane = tid & 31;
  const int wave = tid >> 5;
  const int m0 = blockIdx.x * TM;
  const int n0 = blockIdx.y * TN;

  const int mrow  = lane & 15;
  const int koff8 = (lane >> 4) << 3;   // A: lane-half K offset
  const int nlane = lane & 15;
  const int kb16  = (lane >> 4) << 4;   // unused (B also indexes k with koff8)

  // staging coordinates (16B chunks of 8 bf16)
  const int ar0 = tid >> 2,          ak0 = (tid & 3) << 3;          // A chunk tid
  const int ar1 = (tid + 256) >> 2,  ak1 = ((tid + 256) & 3) << 3;  // A chunk tid+256
  const int br  = tid >> 2,          bk  = (tid & 3) << 3;          // B chunk tid

  auto stage_tile = [&](int buf, int k0) {
    stage16(&A[(size_t)(m0 + ar0) * K + k0 + ak0], &As[buf][ar0 * LDT + ak0]);
    stage16(&A[(size_t)(m0 + ar1) * K + k0 + ak1], &As[buf][ar1 * LDT + ak1]);
    stage16(&Bm[(size_t)(n0 + br) * K + k0 + bk],  &Bs[buf][br * LDT + bk]);
  };

  v8f acc[4] = {};

  stage_tile(0, 0);
  int buf = 0;
  for (int k0 = 0; k0 < K; k0 += TK, buf ^= 1) {
    const bool more = (k0 + TK) < K;
    if (more) stage_tile(buf ^ 1, k0 + TK);     // overlap next tile with compute
    if (more) wait_stage<3>(); else wait_stage<0>();
    __syncthreads();                            // current tile visible to all waves

    // L2 warm-up two tiles ahead (global_prefetch_b8)
    if (k0 + 2 * TK < K) {
      __builtin_prefetch(&A[(size_t)(m0 + (tid >> 1)) * K + k0 + 2 * TK], 0, 1);
      __builtin_prefetch(&Bm[(size_t)(n0 + (tid >> 2)) * K + k0 + 2 * TK], 0, 1);
    }

    // A fragment: 16x32 bf16 per ISA layout
    FragAB a;
#pragma unroll
    for (int v = 0; v < 8; ++v) {
      int k = ((v >> 2) << 4) + koff8 + ((v & 3) << 1);
      a.u[v] = *(const unsigned int*)&As[buf][(wave * 16 + mrow) * LDT + k];
    }
#pragma unroll
    for (int nt = 0; nt < 4; ++nt) {
      FragAB b;
#pragma unroll
      for (int v = 0; v < 8; ++v) {
        int k = ((v >> 2) << 4) + koff8 + ((v & 3) << 1);
        b.u[v] = *(const unsigned int*)&Bs[buf][(nt * 16 + nlane) * LDT + k];
      }
      acc[nt] = __builtin_amdgcn_wmma_f32_16x16x32_bf16(false, a.bf, false, b.bf,
                                                        (short)0, acc[nt], false, false);
    }
    __syncthreads();                            // readers done before buf is re-staged
  }

  const int rbase = m0 + wave * 16 + ((lane >> 4) << 3);
#pragma unroll
  for (int nt = 0; nt < 4; ++nt)
#pragma unroll
    for (int v = 0; v < 8; ++v)
      C[(size_t)(rbase + v) * N + n0 + nt * 16 + (lane & 15)] = acc[nt][v];
}

// ---------------- fused RMSNorm + RoPE (one wave32 per (row, head)) ----------------
__global__ __launch_bounds__(256)
void rmsnorm_rope(const float* __restrict__ in, unsigned short* __restrict__ out,
                  const float* __restrict__ w, const int* __restrict__ pos,
                  int nh, int nrows) {
  int gw   = (int)((blockIdx.x * (size_t)blockDim.x + threadIdx.x) >> 5);
  int lane = threadIdx.x & 31;
  if (gw >= nrows * nh) return;
  int row = gw / nh;
  int h   = gw - row * nh;
  int s   = row & (SEQ - 1);

  const float* p = in + (size_t)row * (nh * DH) + h * DH;
  float t0 = p[lane], t1 = p[lane + 32], t2 = p[lane + 64], t3 = p[lane + 96];
  float ss = t0 * t0 + t1 * t1 + t2 * t2 + t3 * t3;
#pragma unroll
  for (int m = 16; m >= 1; m >>= 1) ss += __shfl_xor(ss, m, 32);
  float r = rsqrtf(ss * (1.0f / DH) + EPSV);
  t0 *= r * w[lane]; t1 *= r * w[lane + 32]; t2 *= r * w[lane + 64]; t3 *= r * w[lane + 96];

  float fp = (float)pos[s];
  float f0 = fp * __expf(-(float)(2 * lane)        * (LOGBASE / (float)DH));
  float f1 = fp * __expf(-(float)(2 * (lane + 32)) * (LOGBASE / (float)DH));
  float c0 = __cosf(f0), s0 = __sinf(f0), c1 = __cosf(f1), s1 = __sinf(f1);
  float o0 = t0 * c0 - t2 * s0;
  float o2 = t2 * c0 + t0 * s0;
  float o1 = t1 * c1 - t3 * s1;
  float o3 = t3 * c1 + t1 * s1;

  unsigned short* q = out + (size_t)row * (nh * DH) + h * DH;
  q[lane] = f2bf(o0); q[lane + 32] = f2bf(o1); q[lane + 64] = f2bf(o2); q[lane + 96] = f2bf(o3);
}

// ---------------- flash attention (causal, GQA) ----------------
// grid = (SEQ/128, HQ, BATCH); 8 waves; wave owns 16 queries; keys in chunks of 32.
// Double-buffered async K/V staging (4 async b128 ops per thread per chunk).
__global__ __launch_bounds__(256)
void attn_fa(const unsigned short* __restrict__ Q, const unsigned short* __restrict__ Kh,
             const unsigned short* __restrict__ Vh, unsigned short* __restrict__ O) {
  __shared__ __align__(16) unsigned short Ks[2][32 * DH];
  __shared__ __align__(16) unsigned short Vs[2][32 * DH];
  __shared__ __align__(16) unsigned short Ps[8][16 * 32];

  const int tid = threadIdx.x, lane = tid & 31, wave = tid >> 5;
  const int qt = blockIdx.x, hq = blockIdx.y, b = blockIdx.z;
  const int hkv = hq >> 1;                 // GQA = 2
  const int q0 = qt * 128;
  const int qbase = q0 + wave * 16;
  const size_t qrs = HQ * DH;              // 2048
  const size_t krs = HKVN * DH;            // 1024

  const unsigned short* Qp = Q  + (size_t)b * SEQ * qrs + hq * DH;
  const unsigned short* Kp = Kh + (size_t)b * SEQ * krs + hkv * DH;
  const unsigned short* Vp = Vh + (size_t)b * SEQ * krs + hkv * DH;

  const int mrow  = lane & 15;
  const int koff8 = (lane >> 4) << 3;
  const int nlane = lane & 15;
  const int kb16  = (lane >> 4) << 4;

  // staging coords: 512 chunks of 8 bf16 per matrix (32 keys x 128 d)
  const int kr0 = tid >> 4,          dc0 = (tid & 15) << 3;
  const int kr1 = (tid + 256) >> 4,  dc1 = ((tid + 256) & 15) << 3;

  auto stage_kv = [&](int buf, int key0) {
    stage16(&Kp[(size_t)(key0 + kr0) * krs + dc0], &Ks[buf][kr0 * DH + dc0]);
    stage16(&Vp[(size_t)(key0 + kr0) * krs + dc0], &Vs[buf][kr0 * DH + dc0]);
    stage16(&Kp[(size_t)(key0 + kr1) * krs + dc1], &Ks[buf][kr1 * DH + dc1]);
    stage16(&Vp[(size_t)(key0 + kr1) * krs + dc1], &Vs[buf][kr1 * DH + dc1]);
  };

  // Q fragments: 4 x (16x32) covering D=128
  FragAB qf[4];
#pragma unroll
  for (int kd = 0; kd < 4; ++kd)
#pragma unroll
    for (int v = 0; v < 8; ++v) {
      int k = kd * 32 + ((v >> 2) << 4) + koff8 + ((v & 3) << 1);
      qf[kd].u[v] = *(const unsigned int*)&Qp[(size_t)(qbase + mrow) * qrs + k];
    }

  v8f o[8] = {};
  float mst[8], lst[8];
#pragma unroll
  for (int v = 0; v < 8; ++v) { mst[v] = -1e30f; lst[v] = 0.f; }

  const int nkc = (q0 + 128) >> 5;
  stage_kv(0, 0);
  int buf = 0;
  for (int kc = 0; kc < nkc; ++kc, buf ^= 1) {
    const int key0 = kc * 32;
    const bool more = (kc + 1) < nkc;
    if (more) stage_kv(buf ^ 1, key0 + 32);     // overlap next chunk with compute
    if (more) wait_stage<4>(); else wait_stage<0>();
    __syncthreads();

    // scores: Q(16x128) x K^T -> two 16x16 tiles
    v8f sc[2] = {};
#pragma unroll
    for (int t = 0; t < 2; ++t)
#pragma unroll
      for (int kd = 0; kd < 4; ++kd) {
        FragAB bk;
#pragma unroll
        for (int v = 0; v < 8; ++v) {
          int dk = kd * 32 + ((v >> 2) << 4) + koff8 + ((v & 3) << 1);
          bk.u[v] = *(const unsigned int*)&Ks[buf][(t * 16 + nlane) * DH + dk];
        }
        sc[t] = __builtin_amdgcn_wmma_f32_16x16x32_bf16(false, qf[kd].bf, false, bk.bf,
                                                        (short)0, sc[t], false, false);
      }

    // scale + causal mask + online softmax
    float rowmax[8];
#pragma unroll
    for (int v = 0; v < 8; ++v) {
      int qrow = qbase + v + ((lane >> 4) << 3);
#pragma unroll
      for (int t = 0; t < 2; ++t) {
        int kg = key0 + t * 16 + (lane & 15);
        float s = sc[t][v] * SCALEV;
        if (kg > qrow) s = -1e30f;
        sc[t][v] = s;
      }
      float rm = fmaxf(sc[0][v], sc[1][v]);
#pragma unroll
      for (int msk = 8; msk >= 1; msk >>= 1) rm = fmaxf(rm, __shfl_xor(rm, msk, 32));
      rowmax[v] = rm;
    }
#pragma unroll
    for (int v = 0; v < 8; ++v) {
      float mnew = fmaxf(mst[v], rowmax[v]);
      float corr = __expf(mst[v] - mnew);
      mst[v] = mnew;
      float p0 = __expf(sc[0][v] - mnew);
      float p1 = __expf(sc[1][v] - mnew);
      sc[0][v] = p0; sc[1][v] = p1;
      float rs = p0 + p1;
#pragma unroll
      for (int msk = 8; msk >= 1; msk >>= 1) rs += __shfl_xor(rs, msk, 32);
      lst[v] = lst[v] * corr + rs;
#pragma unroll
      for (int dt = 0; dt < 8; ++dt) o[dt][v] *= corr;
    }

    // D-layout -> A-layout via per-wave LDS round-trip (DS ops in order per wave)
    unsigned short* pb = &Ps[wave][0];
#pragma unroll
    for (int t = 0; t < 2; ++t)
#pragma unroll
      for (int v = 0; v < 8; ++v) {
        int mr = v + ((lane >> 4) << 3);
        pb[mr * 32 + t * 16 + (lane & 15)] = f2bf(sc[t][v]);
      }
    FragAB pf;
#pragma unroll
    for (int v = 0; v < 8; ++v) {
      int k = ((v >> 2) << 4) + koff8 + ((v & 3) << 1);
      pf.u[v] = *(const unsigned int*)&pb[mrow * 32 + k];
    }
    // O += P(16x32) x V(32x128): 8 output d-tiles
#pragma unroll
    for (int dt = 0; dt < 8; ++dt) {
      FragAB bv;
#pragma unroll
      for (int v = 0; v < 8; ++v) {
        int k = ((v >> 2) << 4) + koff8 + ((v & 3) << 1);
        unsigned int lo = Vs[buf][k * DH + dt * 16 + nlane];
        unsigned int hi = Vs[buf][(k + 1) * DH + dt * 16 + nlane];
        bv.u[v] = lo | (hi << 16);
      }
      o[dt] = __builtin_amdgcn_wmma_f32_16x16x32_bf16(false, pf.bf, false, bv.bf,
                                                      (short)0, o[dt], false, false);
    }
    __syncthreads();                            // readers done before buf is re-staged
  }

  unsigned short* Op = O + (size_t)b * SEQ * qrs + hq * DH;
#pragma unroll
  for (int v = 0; v < 8; ++v) {
    float inv = 1.0f / lst[v];
    int qrow = qbase + v + ((lane >> 4) << 3);
#pragma unroll
    for (int dt = 0; dt < 8; ++dt)
      Op[(size_t)qrow * qrs + dt * 16 + (lane & 15)] = f2bf(o[dt][v] * inv);
  }
}

// ---------------- host-side orchestration ----------------
extern "C" void kernel_launch(void* const* d_in, const int* in_sizes, int n_in,
                              void* d_out, int out_size, void* d_ws, size_t ws_size,
                              hipStream_t stream) {
  (void)in_sizes; (void)n_in; (void)out_size; (void)ws_size;
  const float* x    = (const float*)d_in[0];
  const int*   pos  = (const int*)d_in[1];
  const float* wq   = (const float*)d_in[2];
  const float* wk   = (const float*)d_in[3];
  const float* wv   = (const float*)d_in[4];
  const float* wo   = (const float*)d_in[5];
  const float* qnw  = (const float*)d_in[6];
  const float* knw  = (const float*)d_in[7];
  float* out = (float*)d_out;

  const int M = BATCH * SEQ;             // 4096
  char* w = (char*)d_ws;
  size_t off = 0;
  auto alloc = [&](size_t bytes) { void* p = w + off; off += (bytes + 255) & ~(size_t)255; return p; };

  unsigned short* xh   = (unsigned short*)alloc((size_t)M * HID * 2);        // 16MB (later reused: kh/vh)
  unsigned short* wqh  = (unsigned short*)alloc((size_t)HID * HID * 2);      // 8MB
  unsigned short* wkh  = (unsigned short*)alloc((size_t)(HKVN*DH) * HID * 2);// 4MB
  unsigned short* wvh  = (unsigned short*)alloc((size_t)(HKVN*DH) * HID * 2);// 4MB
  unsigned short* woh  = (unsigned short*)alloc((size_t)HID * HID * 2);      // 8MB
  float* qf = (float*)alloc((size_t)M * HID * 4);                            // 32MB (later reused: attnh)
  float* kf = (float*)alloc((size_t)M * (HKVN*DH) * 4);                      // 16MB
  float* vf = (float*)alloc((size_t)M * (HKVN*DH) * 4);                      // 16MB
  unsigned short* qh = (unsigned short*)alloc((size_t)M * HID * 2);          // 16MB
  // aliases (consumed-before-reuse, stream-ordered):
  unsigned short* kh = xh;                                // 8MB into xh region
  unsigned short* vh = xh + (size_t)M * (HKVN*DH);        // next 8MB of xh region
  unsigned short* attnh = (unsigned short*)qf;            // 16MB into qf region

  auto cdiv = [](int a, int b) { return (a + b - 1) / b; };

  // 1) fp32 -> bf16 for activations + weights (x4 vectorized)
  cvt_f32_bf16<<<cdiv(M * HID, 1024), 256, 0, stream>>>(x, xh, M * HID);
  cvt_f32_bf16<<<cdiv(HID * HID, 1024), 256, 0, stream>>>(wq, wqh, HID * HID);
  cvt_f32_bf16<<<cdiv(HKVN*DH * HID, 1024), 256, 0, stream>>>(wk, wkh, HKVN*DH * HID);
  cvt_f32_bf16<<<cdiv(HKVN*DH * HID, 1024), 256, 0, stream>>>(wv, wvh, HKVN*DH * HID);
  cvt_f32_bf16<<<cdiv(HID * HID, 1024), 256, 0, stream>>>(wo, woh, HID * HID);

  // 2) QKV projections (WMMA GEMM)
  gemm_bf16_f32<<<dim3(M / TM, HID / TN), 256, 0, stream>>>(xh, wqh, qf, M, HID, HID);
  gemm_bf16_f32<<<dim3(M / TM, (HKVN*DH) / TN), 256, 0, stream>>>(xh, wkh, kf, M, HKVN*DH, HID);
  gemm_bf16_f32<<<dim3(M / TM, (HKVN*DH) / TN), 256, 0, stream>>>(xh, wvh, vf, M, HKVN*DH, HID);

  // 3) V -> bf16 ; RMSNorm+RoPE for Q,K -> bf16
  cvt_f32_bf16<<<cdiv(M * HKVN*DH, 1024), 256, 0, stream>>>(vf, vh, M * HKVN*DH);
  rmsnorm_rope<<<cdiv(M * HQ, 8), 256, 0, stream>>>(qf, qh, qnw, pos, HQ, M);
  rmsnorm_rope<<<cdiv(M * HKVN, 8), 256, 0, stream>>>(kf, kh, knw, pos, HKVN, M);

  // 4) causal GQA flash attention (WMMA)
  attn_fa<<<dim3(SEQ / 128, HQ, BATCH), 256, 0, stream>>>(qh, kh, vh, attnh);

  // 5) output projection (WMMA GEMM, fp32 out)
  gemm_bf16_f32<<<dim3(M / TM, HID / TN), 256, 0, stream>>>(attnh, woh, out, M, HID, HID);
}
